// DisenHCN_50199577755802
// MI455X (gfx1250) — matrix operand order repaired and checked by reading, not compiled, not run on hardware
//
#include <hip/hip_runtime.h>
#include <hip/hip_bf16.h>
#include <math.h>

typedef __attribute__((ext_vector_type(16))) _Float16 v16h;
typedef __attribute__((ext_vector_type(8)))  float    v8f;

// ---------------------------------------------------------------------------
// fast transcendentals: CDNA5 has native V_TANH_F32 / V_EXP_F32 (TRANS unit)
// ---------------------------------------------------------------------------
__device__ __forceinline__ float fast_exp2(float x) {
#if __has_builtin(__builtin_amdgcn_exp2f)
  return __builtin_amdgcn_exp2f(x);
#else
  return exp2f(x);
#endif
}

__device__ __forceinline__ float fast_exp(float x) {
  return fast_exp2(x * 1.4426950408889634f);  // e^x = 2^(x*log2 e)
}

__device__ __forceinline__ float fast_tanh(float x) {
#if __has_builtin(__builtin_amdgcn_tanhf)
  return __builtin_amdgcn_tanhf(x);           // native v_tanh_f32
#else
  // branch-free: tanh(x) = 1 - 2/(exp(2x)+1)
  float e = fast_exp2(x * 2.8853900817779268f);
  return 1.0f - 2.0f / (e + 1.0f);
#endif
}

// ---------------------------------------------------------------------------
// init: split user_emb [U,96] into three [U,32] factor buffers (cur + sum)
// ---------------------------------------------------------------------------
__global__ void split_init_kernel(const float* __restrict__ ue,
                                  float* __restrict__ ul, float* __restrict__ ut,
                                  float* __restrict__ ua,
                                  float* __restrict__ sl, float* __restrict__ st,
                                  float* __restrict__ sa, int n) {
  int i = blockIdx.x * blockDim.x + threadIdx.x;
  if (i >= n) return;
  int u = i >> 5, d = i & 31;
  size_t base = (size_t)u * 96;
  float a = ue[base + d];
  float b = ue[base + 32 + d];
  float c = ue[base + 64 + d];
  ul[i] = a; ut[i] = b; ua[i] = c;
  sl[i] = a; st[i] = b; sa[i] = c;
}

__global__ void copy_kernel(const float* __restrict__ s, float* __restrict__ d, int n) {
  int i = blockIdx.x * blockDim.x + threadIdx.x;
  if (i < n) d[i] = s[i];
}

__global__ void accum_kernel(float* __restrict__ dst, const float* __restrict__ src, int n) {
  int i = blockIdx.x * blockDim.x + threadIdx.x;
  if (i < n) dst[i] += src[i];
}

__global__ void scale_store_kernel(const float* __restrict__ src, float* __restrict__ dst,
                                   float sc, int n) {
  int i = blockIdx.x * blockDim.x + threadIdx.x;
  if (i < n) dst[i] = src[i] * sc;
}

__global__ void finalize_users_kernel(const float* __restrict__ sl,
                                      const float* __restrict__ st,
                                      const float* __restrict__ sa,
                                      float* __restrict__ out, int n) {
  int i = blockIdx.x * blockDim.x + threadIdx.x;
  if (i >= n) return;
  int u = i >> 5, d = i & 31;
  const float inv = 1.0f / 3.0f;
  size_t base = (size_t)u * 96;
  out[base + d]      = sl[i] * inv;
  out[base + 32 + d] = st[i] * inv;
  out[base + 64 + d] = sa[i] * inv;
}

// ---------------------------------------------------------------------------
// COO SpMM: y[row, :] += val * x[col, :]  (feature dim = 32 = one wave)
// lane = feature dim -> coalesced 128B gather + coalesced f32 atomic scatter.
// nnz index is wave-uniform: readfirstlane forces the triplet reads onto the
// scalar path (s_load via constant cache) instead of 3 redundant VMEM gathers.
// Triplet stream prefetched ahead (global_prefetch_b8). Accumulators are
// L2-resident (<=25.6MB on 192MB L2), atomics resolve at L2.
// ---------------------------------------------------------------------------
__global__ void spmm_atomic_kernel(const int* __restrict__ rows,
                                   const int* __restrict__ cols,
                                   const float* __restrict__ vals,
                                   const float* __restrict__ x,
                                   float* __restrict__ y,
                                   int ldy, int nnz) {
  int gid = blockIdx.x * blockDim.x + threadIdx.x;
  int e = gid >> 5;
  if (e >= nnz) return;
  int d = gid & 31;
  int eu = __builtin_amdgcn_readfirstlane(e);  // wave-uniform nnz index
  if (eu + 8192 < nnz) {
    __builtin_prefetch(rows + eu + 8192, 0, 0);
    __builtin_prefetch(cols + eu + 8192, 0, 0);
    __builtin_prefetch(vals + eu + 8192, 0, 0);
  }
  int r = rows[eu];
  int c = cols[eu];
  float v = vals[eu];
  atomicAdd(y + (size_t)r * ldy + d, v * x[(size_t)c * 32 + d]);
}

// ---------------------------------------------------------------------------
// Attention fusion with WMMA.
// z: [U, 4, 32]. Per wave: 16 rows of z (4 users x 4 channels) form the A
// matrix (16x32 f16); W1 forms two 32x16 B tiles (N 0..15 and 16..31).
// Two v_wmma_f32_16x16x32_f16 produce h = z @ W1 (16x32, f32 accum).
// Epilogue: tanh(h+b1)*w2 via native v_tanh_f32, row-sum via 16-lane
// shfl_xor reduction, softmax over 4 channels, weighted channel sum.
// ---------------------------------------------------------------------------
__global__ void attn_fuse_wmma_kernel(const float* __restrict__ z,
                                      const float* __restrict__ W1g,
                                      const float* __restrict__ b1g,
                                      const float* __restrict__ w2g,
                                      float* __restrict__ u_out,
                                      float* __restrict__ u_sum,
                                      int nUsers) {
  __shared__ _Float16 sW1[32 * 32];
  __shared__ float sB1[32];
  __shared__ float sW2[32];

  int tid = threadIdx.x;
  for (int i = tid; i < 32 * 32; i += blockDim.x) sW1[i] = (_Float16)W1g[i];
  if (tid < 32) { sB1[tid] = b1g[tid]; sW2[tid] = w2g[tid]; }
  __syncthreads();

  int wave = tid >> 5;
  int lane = tid & 31;
  int u0 = (blockIdx.x * (blockDim.x >> 5) + wave) * 4;  // 4 users per wave
  if (u0 >= nUsers) return;  // wave-uniform: EXEC stays all-1s for WMMA

  int m  = lane & 15;   // A: row M; B/D: column N (within half)
  int hi = lane >> 4;   // lane half selects K sub-range / M sub-range

  // A fragment: 16x32 f16; documented CDNA5 layout:
  // lanes 0-15  hold K = 0..7 (elems 0..7) and 16..23 (elems 8..15)
  // lanes 16-31 hold K = 8..15            and 24..31
  v16h a;
  {
    const float* zr = z + (size_t)(u0 * 4 + m) * 32;
#pragma unroll
    for (int j = 0; j < 16; ++j) {
      int k = hi * 8 + (j & 7) + (j >> 3) * 16;
      a[j] = (_Float16)zr[k];
    }
  }

  // B fragments: 32x16 f16 per N-half; lane holds column N = m,
  // K = hi*8 .. hi*8+7 (elems 0..7) and 16+hi*8 .. 16+hi*8+7 (elems 8..15).
  v16h b0, b1h;
#pragma unroll
  for (int j = 0; j < 16; ++j) {
    int k = hi * 8 + (j & 7) + (j >> 3) * 16;
    b0[j]  = sW1[k * 32 + m];
    b1h[j] = sW1[k * 32 + 16 + m];
  }

  v8f c0 = {}, c1 = {};
  c0 = __builtin_amdgcn_wmma_f32_16x16x32_f16(false, a, false, b0,
                                              (short)0, c0, false, false);
  c1 = __builtin_amdgcn_wmma_f32_16x16x32_f16(false, a, false, b1h,
                                              (short)0, c1, false, false);

  // D layout: VGPR r, lane -> M = hi*8 + r, N = m (c0) / 16+m (c1)
  float bb0 = sB1[m], bb1 = sB1[16 + m];
  float ww0 = sW2[m], ww1 = sW2[16 + m];
  float s[8];
#pragma unroll
  for (int r = 0; r < 8; ++r) {
    float t0 = fast_tanh(c0[r] + bb0) * ww0;
    float t1 = fast_tanh(c1[r] + bb1) * ww1;
    s[r] = t0 + t1;
  }
  // Row-sum over N: reduce across the 16 lanes of this half-wave.
#pragma unroll
  for (int r = 0; r < 8; ++r) {
#pragma unroll
    for (int off = 1; off < 16; off <<= 1) s[r] += __shfl_xor(s[r], off, 16);
  }

  // Each lane half covers rows M = hi*8 .. hi*8+7 = users (u0 + hi*2 + g),
  // channels c = M & 3. Softmax over the 4 channels, then weighted sum of z.
#pragma unroll
  for (int g = 0; g < 2; ++g) {
    int u = u0 + hi * 2 + g;
    float w0 = s[g * 4 + 0], w1 = s[g * 4 + 1];
    float w2v = s[g * 4 + 2], w3 = s[g * 4 + 3];
    float mx = fmaxf(fmaxf(w0, w1), fmaxf(w2v, w3));
    float e0 = fast_exp(w0 - mx), e1 = fast_exp(w1 - mx);
    float e2 = fast_exp(w2v - mx), e3 = fast_exp(w3 - mx);
    float inv = 1.0f / (e0 + e1 + e2 + e3);
    const float* zu = z + (size_t)u * 4 * 32;
#pragma unroll
    for (int dh = 0; dh < 2; ++dh) {
      int d = m + dh * 16;
      float o = (e0 * zu[0 * 32 + d] + e1 * zu[1 * 32 + d] +
                 e2 * zu[2 * 32 + d] + e3 * zu[3 * 32 + d]) * inv;
      u_out[(size_t)u * 32 + d] = o;
      u_sum[(size_t)u * 32 + d] += o;
    }
  }
}

// ---------------------------------------------------------------------------
// host side
// ---------------------------------------------------------------------------
extern "C" void kernel_launch(void* const* d_in, const int* in_sizes, int n_in,
                              void* d_out, int out_size, void* d_ws, size_t ws_size,
                              hipStream_t stream) {
  const float* user_emb = (const float*)d_in[0];
  const float* loc_emb  = (const float*)d_in[1];
  const float* time_emb = (const float*)d_in[2];
  const float* act_emb  = (const float*)d_in[3];
  // W1/b1/w2 per factor: l=4..6, t=7..9, a=10..12
  const float* W1[3] = { (const float*)d_in[4], (const float*)d_in[7], (const float*)d_in[10] };
  const float* B1[3] = { (const float*)d_in[5], (const float*)d_in[8], (const float*)d_in[11] };
  const float* W2[3] = { (const float*)d_in[6], (const float*)d_in[9], (const float*)d_in[12] };

  const int U = in_sizes[0] / 96;
  const int L = in_sizes[1] / 32;
  const int T = in_sizes[2] / 32;
  const int A = in_sizes[3] / 32;
  const int NNZ = in_sizes[13];

  // sparse input base indices (rows, cols, vals consecutive)
  const int iNormL = 13, iNormT = 16, iNormA = 19, iNormLT = 22, iNormLA = 25,
            iNormTA = 28, iNormLTA = 31, iVtoeL = 34, iVtoeT = 37, iVtoeA = 40;

  // workspace layout (floats)
  float* ws = (float*)d_ws;
  const size_t U32 = (size_t)U * 32;
  float* u_cur[3] = { ws, ws + U32, ws + 2 * U32 };
  float* u_sum[3] = { ws + 3 * U32, ws + 4 * U32, ws + 5 * U32 };
  float* z = ws + 6 * U32;                       // [U,4,32]
  float* l_cur = z + (size_t)U * 128;
  float* t_cur = l_cur + (size_t)L * 32;
  float* a_cur = t_cur + (size_t)T * 32;
  float* l_sum = a_cur + (size_t)A * 32;
  float* t_sum = l_sum + (size_t)L * 32;
  float* a_sum = t_sum + (size_t)T * 32;
  float* e_cur[3] = { l_cur, t_cur, a_cur };
  float* e_sum[3] = { l_sum, t_sum, a_sum };
  const long e_n[3] = { (long)L * 32, (long)T * 32, (long)A * 32 };

  auto cdiv = [](long a, long b) { return (unsigned)((a + b - 1) / b); };

  auto spmm = [&](int idx, const float* x, float* y, int ldy) {
    spmm_atomic_kernel<<<cdiv((long)NNZ * 32, 256), 256, 0, stream>>>(
        (const int*)d_in[idx], (const int*)d_in[idx + 1],
        (const float*)d_in[idx + 2], x, y, ldy, NNZ);
  };

  // ---- init: layer-0 snapshots seed the running sums ----
  split_init_kernel<<<cdiv((long)U32, 256), 256, 0, stream>>>(
      user_emb, u_cur[0], u_cur[1], u_cur[2], u_sum[0], u_sum[1], u_sum[2], (int)U32);
  copy_kernel<<<cdiv(e_n[0], 256), 256, 0, stream>>>(loc_emb,  l_sum, (int)e_n[0]);
  copy_kernel<<<cdiv(e_n[1], 256), 256, 0, stream>>>(time_emb, t_sum, (int)e_n[1]);
  copy_kernel<<<cdiv(e_n[2], 256), 256, 0, stream>>>(act_emb,  a_sum, (int)e_n[2]);

  // SpMM set per factor (4 hyperedge types each) and vertex->edge matrix
  const int setF[3][4] = { { iNormL, iNormLT, iNormLA, iNormLTA },
                           { iNormT, iNormLT, iNormTA, iNormLTA },
                           { iNormA, iNormLA, iNormTA, iNormLTA } };
  const int vtoe[3] = { iVtoeL, iVtoeT, iVtoeA };

  const unsigned attnGrid = cdiv(U, 32);  // 8 waves/block * 4 users/wave

  for (int layer = 0; layer < 2; ++layer) {
    for (int f = 0; f < 3; ++f) {
      // z[:, c, :] = norm_c @ u_f  (4 channels, row stride 128 into z)
      hipMemsetAsync(z, 0, (size_t)U * 128 * sizeof(float), stream);
      for (int c = 0; c < 4; ++c) spmm(setF[f][c], u_cur[f], z + c * 32, 128);
      // edge update uses the pre-fusion u_f
      hipMemsetAsync(e_cur[f], 0, (size_t)e_n[f] * sizeof(float), stream);
      spmm(vtoe[f], u_cur[f], e_cur[f], 32);
      // inter-type attention fusion (WMMA) -> new u_f, accumulated into sum
      attn_fuse_wmma_kernel<<<attnGrid, 256, 0, stream>>>(
          z, W1[f], B1[f], W2[f], u_cur[f], u_sum[f], U);
      accum_kernel<<<cdiv(e_n[f], 256), 256, 0, stream>>>(e_sum[f], e_cur[f], (int)e_n[f]);
    }
  }

  // ---- finalize: means over the 3 snapshots ----
  float* out = (float*)d_out;
  finalize_users_kernel<<<cdiv((long)U32, 256), 256, 0, stream>>>(
      u_sum[0], u_sum[1], u_sum[2], out, (int)U32);
  float* out_e = out + (size_t)U * 96;
  const float inv3 = 1.0f / 3.0f;
  for (int f = 0; f < 3; ++f) {
    scale_store_kernel<<<cdiv(e_n[f], 256), 256, 0, stream>>>(
        e_sum[f], out_e, inv3, (int)e_n[f]);
    out_e += e_n[f];
  }
}
